// ASTCODA_45372034515119
// MI455X (gfx1250) — matrix operand order repaired and checked by reading, not compiled, not run on hardware
//
#include <hip/hip_runtime.h>
#include <hip/hip_bf16.h>
#include <math.h>

// Problem constants (from reference)
#define B_    64
#define S_    2048
#define E_    128
#define K_    5
#define F_    512
#define D_    16
#define C_    10
#define PADW  (K_ - 1)            // 4
#define T_    (S_ + 2 * PADW)     // 2056
#define L_    (S_ + K_ - 1)       // 2052
#define T_PAD 2184                // covers l0_max(2048)+127+4 = 2179, 8-row slack
#define KE_   (K_ * E_)           // 640
#define BN_EPS 1e-5f

typedef __attribute__((ext_vector_type(16))) _Float16 v16h;
typedef __attribute__((ext_vector_type(8)))  _Float16 v8h;
typedef __attribute__((ext_vector_type(8)))  float    v8f;

#define LA_STRIDE 40   // 32 data halves + 8 pad -> 80B row stride, conflict-free b128 frag loads
#define LB_STRIDE 40

// ---------------------------------------------------------------------------
// Kernel 1: conv_w (f32 [F, K*E]) -> f16
// ---------------------------------------------------------------------------
__global__ __launch_bounds__(256) void wconv_kernel(const float* __restrict__ w,
                                                    _Float16* __restrict__ w16, int n) {
  int i = blockIdx.x * 256 + threadIdx.x;
  if (i < n) w16[i] = (_Float16)w[i];
}

// ---------------------------------------------------------------------------
// Kernel 2: gather + convert embeddings -> f16 [B, T_PAD, E], zero tail/pads.
// 2 elements per thread: float2 load -> packed 2 x f16 store (full-width waves)
// ---------------------------------------------------------------------------
__global__ __launch_bounds__(256) void emb_gather_kernel(const int* __restrict__ data,
                                                         const float* __restrict__ emb_table,
                                                         _Float16* __restrict__ emb16) {
  const int EP = E_ / 2;  // 64 f16-pairs per row
  size_t gid = (size_t)blockIdx.x * 256 + threadIdx.x;
  const size_t totalPairs = (size_t)B_ * T_PAD * EP;
  if (gid >= totalPairs) return;
  int ep = (int)(gid % EP);
  size_t rt = gid / EP;
  int t = (int)(rt % T_PAD);
  int b = (int)(rt / T_PAD);
  float vx = 0.0f, vy = 0.0f;
  if (t < T_) {
    int tok = 0;  // ConstantPad1d value == PAD_IDX; emb_table[0] is the zero row
    if (t >= PADW && t < PADW + S_) tok = data[(size_t)b * S_ + (t - PADW)];
    const float2 v = *(const float2*)&emb_table[(size_t)tok * E_ + 2 * ep];
    vx = v.x;
    vy = v.y;
  }
  union { _Float16 h[2]; unsigned int u; } pk;
  pk.h[0] = (_Float16)vx;
  pk.h[1] = (_Float16)vy;
  *(unsigned int*)&emb16[gid * 2] = pk.u;
}

// ---------------------------------------------------------------------------
// Kernel 3: batched GEMM features[b,f,l] = window(emb) x conv_w^T via WMMA,
// fused conv bias + BatchNorm(eval) + ReLU.
// Grid: (F/64, ceil(L/128), B); 256 threads = 8 waves, wave tile 32x32.
// Register-staged, LDS double-buffered pipeline: loads for chunk c+1 issue
// before the barrier and retire into LDS after the chunk-c WMMAs.
// ---------------------------------------------------------------------------
__global__ __launch_bounds__(256) void conv_gemm_kernel(
    const _Float16* __restrict__ emb16, const _Float16* __restrict__ w16,
    const float* __restrict__ conv_b, const float* __restrict__ gamma,
    const float* __restrict__ beta, const float* __restrict__ rmean,
    const float* __restrict__ rvar, float* __restrict__ features,
    float* __restrict__ act) {
  __shared__ _Float16 lA[2][128 * LA_STRIDE];  // A chunk: 128 rows(l) x 32 halves(k)
  __shared__ _Float16 lB[2][64 * LB_STRIDE];   // B chunk: 64 cols(f) x 32 halves(k)

  const int b  = blockIdx.z;
  const int l0 = blockIdx.y * 128;
  const int f0 = blockIdx.x * 64;
  const int tid  = threadIdx.x;
  const int lane = tid & 31;
  const int wv   = tid >> 5;
  const int wl   = wv & 3;   // 4 waves along L (32 rows each)
  const int wf   = wv >> 2;  // 2 waves along F (32 cols each)

  const _Float16* Abase = emb16 + (size_t)b * T_PAD * E_;

  // Per-thread stage slots.
  // A: 128 rows x 64B -> 2 threads/row, 32B each (two v8h).
  const int arow = tid >> 1;
  const int aoff = (tid & 1) * 16;                  // half offset within row (0 or 16)
  const int aldso = arow * LA_STRIDE + aoff;
  // B: 64 f-rows x 64B -> 4 threads/row, 16B each (one v8h). conv_w row-major
  // over K*E means K is contiguous per f -> B arrives pre-"transposed".
  const int bfr = tid >> 2;
  const int bsg = tid & 3;
  const int bldso = bfr * LB_STRIDE + bsg * 8;

  // Fragment LDS offsets (constant across chunks).
  const int r    = lane & 15;
  const int kbA  = (lane >> 4) * 8;
  const int ra0  = (wl * 32 + r) * LA_STRIDE + kbA;
  const int ra1  = (wl * 32 + 16 + r) * LA_STRIDE + kbA;
  const int kbB  = (lane >> 4) * 8;
  const int cb0  = (wf * 32 + (lane & 15)) * LB_STRIDE + kbB;
  const int cb1  = (wf * 32 + 16 + (lane & 15)) * LB_STRIDE + kbB;

  v8f acc[2][2] = {};
  v8h rA0, rA1, rB;

  // --- load chunk 0 into registers, store to buffer 0 ---
  {
    const _Float16* gA = Abase + (size_t)(l0 + arow) * E_ + aoff;  // c=0: tshift=0, e0=0
    rA0 = ((const v8h*)gA)[0];
    rA1 = ((const v8h*)gA)[1];
    const _Float16* gB = w16 + (size_t)(f0 + bfr) * KE_ + bsg * 8;
    rB = *(const v8h*)gB;
  }
  *(v8h*)&lA[0][aldso]     = rA0;
  *(v8h*)&lA[0][aldso + 8] = rA1;
  *(v8h*)&lB[0][bldso]     = rB;

#pragma unroll 2
  for (int c = 0; c < 20; ++c) {
    // Issue global loads for chunk c+1 (retired into LDS after this chunk's WMMAs)
    if (c < 19) {
      const int nc = c + 1;
      const _Float16* gA = Abase + (size_t)(l0 + arow + (nc >> 2)) * E_ +
                           (nc & 3) * 32 + aoff;
      rA0 = ((const v8h*)gA)[0];
      rA1 = ((const v8h*)gA)[1];
      const _Float16* gB = w16 + (size_t)(f0 + bfr) * KE_ + nc * 32 + bsg * 8;
      rB = *(const v8h*)gB;
      if (c < 18) {
        const int pc = c + 2;  // prefetch two chunks ahead (global_prefetch_b8)
        __builtin_prefetch(Abase + (size_t)(l0 + arow + (pc >> 2)) * E_ +
                               (pc & 3) * 32 + aoff, 0, 0);
      }
    }

    __syncthreads();  // buffer (c&1) fully written; prior reads of (c+1)&1 done

    const _Float16* bufA = lA[c & 1];
    const _Float16* bufB = lB[c & 1];

    // A fragments (16-bit A 16x32 layout): lane<16 holds K{0..7,16..23} of
    // row lane; lane>=16 holds K{8..15,24..31} of row lane-16.
    v8h a0lo = *(const v8h*)&bufA[ra0];
    v8h a0hi = *(const v8h*)&bufA[ra0 + 16];
    v8h a1lo = *(const v8h*)&bufA[ra1];
    v8h a1hi = *(const v8h*)&bufA[ra1 + 16];
    v16h a0 = __builtin_shufflevector(a0lo, a0hi, 0,1,2,3,4,5,6,7,8,9,10,11,12,13,14,15);
    v16h a1 = __builtin_shufflevector(a1lo, a1hi, 0,1,2,3,4,5,6,7,8,9,10,11,12,13,14,15);

    // B fragments (B 32x16): lane<16 -> N=lane, K{0..7,16..23}; lane>=16 -> N=lane-16, K{8..15,24..31}
    v8h b0lo = *(const v8h*)&bufB[cb0];
    v8h b0hi = *(const v8h*)&bufB[cb0 + 16];
    v8h b1lo = *(const v8h*)&bufB[cb1];
    v8h b1hi = *(const v8h*)&bufB[cb1 + 16];
    v16h bm0 = __builtin_shufflevector(b0lo, b0hi, 0,1,2,3,4,5,6,7,8,9,10,11,12,13,14,15);
    v16h bm1 = __builtin_shufflevector(b1lo, b1hi, 0,1,2,3,4,5,6,7,8,9,10,11,12,13,14,15);

    acc[0][0] = __builtin_amdgcn_wmma_f32_16x16x32_f16(false, a0, false, bm0,
                                                       (short)0, acc[0][0], false, false);
    acc[0][1] = __builtin_amdgcn_wmma_f32_16x16x32_f16(false, a0, false, bm1,
                                                       (short)0, acc[0][1], false, false);
    acc[1][0] = __builtin_amdgcn_wmma_f32_16x16x32_f16(false, a1, false, bm0,
                                                       (short)0, acc[1][0], false, false);
    acc[1][1] = __builtin_amdgcn_wmma_f32_16x16x32_f16(false, a1, false, bm1,
                                                       (short)0, acc[1][1], false, false);

    // Retire staged registers into the other buffer for chunk c+1
    if (c < 19) {
      _Float16* nA = lA[(c + 1) & 1];
      _Float16* nB = lB[(c + 1) & 1];
      *(v8h*)&nA[aldso]     = rA0;
      *(v8h*)&nA[aldso + 8] = rA1;
      *(v8h*)&nB[bldso]     = rB;
    }
  }

  // Epilogue: + conv_b, BN(eval) + ReLU. D layout: VGPR i -> M = i (+8 if lane>=16), N = lane&15.
  for (int tj = 0; tj < 2; ++tj) {
    const int f = f0 + wf * 32 + tj * 16 + (lane & 15);
    const float cb  = conv_b[f];
    const float sc  = gamma[f] * rsqrtf(rvar[f] + BN_EPS);
    const float sh  = beta[f] - rmean[f] * sc;
    for (int ti = 0; ti < 2; ++ti) {
      const int lbase = l0 + wl * 32 + ti * 16 + ((lane >> 4) << 3);
      v8f& A = acc[ti][tj];
      for (int i = 0; i < 8; ++i) {
        const int l = lbase + i;
        if (l < L_) {
          const float feat = A[i] + cb;
          const size_t idx = ((size_t)b * F_ + f) * (size_t)L_ + l;
          __builtin_nontemporal_store(feat, &features[idx]);  // streamed, never re-read
          const float a = feat * sc + sh;
          act[idx] = a > 0.0f ? a : 0.0f;
        }
      }
    }
  }
}

// ---------------------------------------------------------------------------
// Kernel 4: per-batch softmax attention + domain-routed linear.
// One block per batch sample; 256 threads.
// ---------------------------------------------------------------------------
__global__ __launch_bounds__(256) void attn_kernel(
    const int* __restrict__ domain, const float* __restrict__ dom_table,
    const float* __restrict__ fc_w, const float* __restrict__ fc_b,
    const float* __restrict__ act, float* __restrict__ out_domains,
    float* __restrict__ out_logits) {
  __shared__ float dom_s[F_];
  __shared__ float p[L_];
  __shared__ float red[256];
  __shared__ float res[F_];

  const int b = blockIdx.x;
  const int tid = threadIdx.x;
  const int lane = tid & 31;
  const int wv = tid >> 5;
  const int dm = domain[b];

  for (int f = tid; f < F_; f += 256) {
    float v = dom_table[dm * F_ + f];
    dom_s[f] = v;
    out_domains[b * F_ + f] = v;
  }
  __syncthreads();

  const float* actb = act + (size_t)b * F_ * (size_t)L_;

  // scores[l] = sum_f dom[f] * act[f][l]  (coalesced over l)
  float sc[9];
#pragma unroll
  for (int j = 0; j < 9; ++j) sc[j] = 0.0f;
  for (int f = 0; f < F_; ++f) {
    const float d = dom_s[f];
    const float* row = actb + (size_t)f * L_;
#pragma unroll
    for (int j = 0; j < 9; ++j) {
      const int l = tid + j * 256;
      if (l < L_) sc[j] = fmaf(d, row[l], sc[j]);
    }
  }
  float mymax = -INFINITY;
#pragma unroll
  for (int j = 0; j < 9; ++j) {
    const int l = tid + j * 256;
    if (l < L_) { p[l] = sc[j]; mymax = fmaxf(mymax, sc[j]); }
  }
  red[tid] = mymax;
  __syncthreads();
  for (int s = 128; s > 0; s >>= 1) {
    if (tid < s) red[tid] = fmaxf(red[tid], red[tid + s]);
    __syncthreads();
  }
  const float mx = red[0];
  __syncthreads();
  float mysum = 0.0f;
#pragma unroll
  for (int j = 0; j < 9; ++j) {
    const int l = tid + j * 256;
    if (l < L_) {
      const float e = __expf(p[l] - mx);
      p[l] = e;
      mysum += e;
    }
  }
  red[tid] = mysum;
  __syncthreads();
  for (int s = 128; s > 0; s >>= 1) {
    if (tid < s) red[tid] += red[tid + s];
    __syncthreads();
  }
  const float sinv = 1.0f / red[0];
  __syncthreads();

  // result[f] = sum_l softmax(l) * act[f][l]; 8 waves x 64 f each, lanes over l
  for (int fi = 0; fi < F_ / 8; ++fi) {
    const int f = wv * (F_ / 8) + fi;
    const float* row = actb + (size_t)f * L_;
    float s = 0.0f;
    for (int l = lane; l < L_; l += 32) s = fmaf(p[l], row[l], s);
#pragma unroll
    for (int m = 16; m; m >>= 1) s += __shfl_xor(s, m, 32);
    if (lane == 0) res[f] = s * sinv;
  }
  __syncthreads();

  // logits[c] = result . fc_w[dm, c, :] + fc_b[dm, c]   (tiny; wave 0)
  if (wv == 0) {
    for (int c = 0; c < C_; ++c) {
      const float* wrow = fc_w + ((size_t)dm * C_ + c) * F_;
      float s = 0.0f;
      for (int f = lane; f < F_; f += 32) s = fmaf(res[f], wrow[f], s);
#pragma unroll
      for (int m = 16; m; m >>= 1) s += __shfl_xor(s, m, 32);
      if (lane == 0) out_logits[b * C_ + c] = s + fc_b[dm * C_ + c];
    }
  }
}

// ---------------------------------------------------------------------------
extern "C" void kernel_launch(void* const* d_in, const int* in_sizes, int n_in,
                              void* d_out, int out_size, void* d_ws, size_t ws_size,
                              hipStream_t stream) {
  const int*   data      = (const int*)d_in[0];
  const int*   domain    = (const int*)d_in[1];
  const float* emb_table = (const float*)d_in[2];
  const float* conv_w    = (const float*)d_in[3];
  const float* conv_b    = (const float*)d_in[4];
  const float* gamma     = (const float*)d_in[5];
  const float* beta      = (const float*)d_in[6];
  const float* run_mean  = (const float*)d_in[7];
  const float* run_var   = (const float*)d_in[8];
  const float* dom_table = (const float*)d_in[9];
  const float* fc_w      = (const float*)d_in[10];
  const float* fc_b      = (const float*)d_in[11];

  float* out      = (float*)d_out;
  float* features = out;                                   // [B, F, L]
  float* act      = features + (size_t)B_ * F_ * L_;       // [B, F, L]
  float* out_dom  = act + (size_t)B_ * F_ * L_;            // [B, F]
  float* out_log  = out_dom + (size_t)B_ * F_;             // [B, C]

  _Float16* emb16 = (_Float16*)d_ws;                       // [B, T_PAD, E] ~35.8MB
  _Float16* w16   = emb16 + (size_t)B_ * T_PAD * E_;       // [F, K*E] ~0.64MB

  // 1) conv_w -> f16
  const int nw = F_ * KE_;
  wconv_kernel<<<(nw + 255) / 256, 256, 0, stream>>>(conv_w, w16, nw);

  // 2) gather + convert embeddings (2 elems/thread)
  const size_t npairs = (size_t)B_ * T_PAD * (E_ / 2);
  emb_gather_kernel<<<(unsigned)((npairs + 255) / 256), 256, 0, stream>>>(data, emb_table, emb16);

  // 3) WMMA GEMM + fused bias/BN/ReLU
  dim3 grid(F_ / 64, (L_ + 127) / 128, B_);  // (8, 17, 64)
  conv_gemm_kernel<<<grid, 256, 0, stream>>>(emb16, w16, conv_b, gamma, beta,
                                             run_mean, run_var, features, act);

  // 4) attention + routed linear
  attn_kernel<<<B_, 256, 0, stream>>>(domain, dom_table, fc_w, fc_b, act,
                                      out_dom, out_log);
}
